// WindowTopKAttention_4690104287953
// MI455X (gfx1250) — compile-verified, matrix operand order checked by reading, not run on hardware
//
#include <hip/hip_runtime.h>
#include <hip/hip_bf16.h>

typedef __attribute__((ext_vector_type(16))) _Float16 v16h;
typedef __attribute__((ext_vector_type(8)))  _Float16 v8h;
typedef __attribute__((ext_vector_type(8)))  float    v8f;
typedef __attribute__((ext_vector_type(4)))  float    v4f;

#define BPW   256   // number of windows (B * NW)
#define NTOK  256   // tokens per window (16x16)
#define CDIM  256
#define NHEAD 8
#define HDIM  32
#define KTOP  8
#define BB    4
#define LSEQ  512   // NW * TOPK

// ---------------- WMMA helpers ----------------

__device__ __forceinline__ v8f wmma16(v16h a, v16h b, v8f c) {
  return __builtin_amdgcn_wmma_f32_16x16x32_f16(false, a, false, b, (short)0, c,
                                                false, false);
}

// A/B fragment (16x32 f16, row-major source with contiguous K).
// Per ISA: lanes 0-15 hold rows M=0..15, k pattern {c0..c0+7, c0+16..c0+23},
// c0 = 8 for lanes 16-31.  Two 16-byte loads per lane.
__device__ __forceinline__ v16h frag_ld(const _Float16* __restrict__ base,
                                        int stride, int lane) {
  const int m  = lane & 15;
  const int c0 = (lane >> 4) << 3;
  const _Float16* r = base + (long)m * stride + c0;
  v8h lo = *(const v8h*)(r);
  v8h hi = *(const v8h*)(r + 16);
  v16h f;
#pragma unroll
  for (int i = 0; i < 8; ++i) { f[i] = lo[i]; f[i + 8] = hi[i]; }
  return f;
}

// Same fragment but sourced from an LDS panel with row stride 32 halves.
__device__ __forceinline__ v16h frag_ld_lds(const _Float16* bs, int row0,
                                            int lane) {
  const int m  = lane & 15;
  const int c0 = (lane >> 4) << 3;
  const _Float16* r = bs + (row0 + m) * 32 + c0;
  v8h lo = *(const v8h*)(r);
  v8h hi = *(const v8h*)(r + 16);
  v16h f;
#pragma unroll
  for (int i = 0; i < 8; ++i) { f[i] = lo[i]; f[i + 8] = hi[i]; }
  return f;
}

__device__ __forceinline__ float wave_sum(float v) {
#pragma unroll
  for (int d = 16; d >= 1; d >>= 1) v += __shfl_xor(v, d, 32);
  return v;
}
__device__ __forceinline__ float wave_max(float v) {
#pragma unroll
  for (int d = 16; d >= 1; d >>= 1) v = fmaxf(v, __shfl_xor(v, d, 32));
  return v;
}

// ---------------- block-cooperative WMMA GEMM ----------------
// Block (256 threads, 8 waves) computes a 128x64 macro-tile of
// out[M,Nc] = A[M,K](f16) * W + bias, with Bt[n][k] (f16, transposed weight).
// The 64-column B panel is streamed through a double-buffered LDS stage
// using CDNA5 async global->LDS copies (ASYNCcnt + s_wait_asynccnt), so the
// fetch of chunk k+1 overlaps the 4 WMMAs of chunk k.  Each wave owns 16 rows
// and reuses one A fragment across 4 column tiles.
__global__ __launch_bounds__(256) void k_gemm(
    const _Float16* __restrict__ A, const _Float16* __restrict__ Bt,
    const float* __restrict__ bias, float* __restrict__ outF32,
    _Float16* __restrict__ outF16, _Float16* __restrict__ outT,
    int M, int Nc, int K, long bWinStride, int winM, int actRelu) {
  __shared__ _Float16 Bs[2][64][32];  // 2 x 4 KB double buffer
  const int n0g = blockIdx.x * 64;    // column panel base
  const int mBase = blockIdx.y * 128; // row group base
  const int w = threadIdx.x >> 5, lane = threadIdx.x & 31;
  const int m0 = mBase + w * 16;
  const _Float16* Bw =
      Bt + (bWinStride ? (long)(mBase / winM) * bWinStride : 0);

  // cooperative copy map: thread t moves 16B of the 64x32 chunk
  const int rn = threadIdx.x >> 2;        // Bt row within panel (0..63)
  const int rp = (threadIdx.x & 3) << 3;  // half offset 0,8,16,24
  const unsigned ldsBase =
      (unsigned)(unsigned long long)(const void*)&Bs[0][rn][rp];
  const _Float16* gRow = Bw + (long)(n0g + rn) * K + rp;

  const int nchunks = K >> 5;
  {
    unsigned long long ga = (unsigned long long)(gRow);
    unsigned la = ldsBase;
    asm volatile("global_load_async_to_lds_b128 %0, %1, off"
                 :: "v"(la), "v"(ga) : "memory");
  }

  v8f acc0 = {}, acc1 = {}, acc2 = {}, acc3 = {};
  for (int c = 0; c < nchunks; ++c) {
    const int buf = c & 1;
    if (c + 1 < nchunks) {
      unsigned long long ga = (unsigned long long)(gRow + (c + 1) * 32);
      unsigned la = ldsBase + ((c + 1) & 1) * 4096u;
      asm volatile("global_load_async_to_lds_b128 %0, %1, off"
                   :: "v"(la), "v"(ga) : "memory");
      asm volatile("s_wait_asynccnt 0x1" ::: "memory");
    } else {
      asm volatile("s_wait_asynccnt 0x0" ::: "memory");
    }
    __syncthreads();  // chunk c resident in LDS for all waves
    v16h a = frag_ld(A + (long)m0 * K + c * 32, K, lane);
    const _Float16* bs = &Bs[buf][0][0];
    v16h b0 = frag_ld_lds(bs, 0, lane);
    v16h b1 = frag_ld_lds(bs, 16, lane);
    v16h b2 = frag_ld_lds(bs, 32, lane);
    v16h b3 = frag_ld_lds(bs, 48, lane);
    acc0 = wmma16(a, b0, acc0);
    acc1 = wmma16(a, b1, acc1);
    acc2 = wmma16(a, b2, acc2);
    acc3 = wmma16(a, b3, acc3);
    __syncthreads();  // reads done before buffer is overwritten
  }

  const int n = lane & 15, hb = lane >> 4;
  v8f accs[4] = {acc0, acc1, acc2, acc3};
#pragma unroll
  for (int ctw = 0; ctw < 4; ++ctw) {
    int ncol = n0g + ctw * 16 + n;
    float bv = bias ? bias[ncol] : 0.f;
    float dv[8];
#pragma unroll
    for (int r = 0; r < 8; ++r) {
      float d = accs[ctw][r] + bv;
      if (actRelu) d = fmaxf(d, 0.f);
      dv[r] = d;
    }
    if (outF32) {
#pragma unroll
      for (int r = 0; r < 8; ++r)
        outF32[(long)(m0 + r + (hb << 3)) * Nc + ncol] = dv[r];
    }
    if (outF16) {
#pragma unroll
      for (int r = 0; r < 8; ++r)
        outF16[(long)(m0 + r + (hb << 3)) * Nc + ncol] = (_Float16)dv[r];
    }
    if (outT) {  // per-window transposed: outT[win][col][rowInWin]
      v8h pk;
#pragma unroll
      for (int r = 0; r < 8; ++r) pk[r] = (_Float16)dv[r];
      int win = m0 / winM, mw = (m0 % winM) + (hb << 3);
      *(v8h*)(outT + ((long)win * Nc + ncol) * winM + mw) = pk;
    }
  }
}

// ---------------- fused per-head attention ----------------
// One wave owns (window bp, head h, 16-row tile m0):
//   S[16,256] = qh * kh^T (WMMA)  -> in-wave softmax (scaled) in LDS (f32)
//   out[16,32] = S * vh           (WMMA, A rebuilt f16 from LDS)
__global__ __launch_bounds__(128) void k_headattn(
    const _Float16* __restrict__ q16, const _Float16* __restrict__ k16,
    const _Float16* __restrict__ vt16, _Float16* __restrict__ out16) {
  __shared__ float S[4][16][256];  // 64 KB
  const int w = threadIdx.x >> 5;
  const int lane = threadIdx.x & 31;
  int task = blockIdx.x * 4 + w;   // [0, 256*8*16)
  int bp = task >> 7;
  int rem = task & 127;
  int h = rem >> 4;
  int m0 = (rem & 15) << 4;
  const int n = lane & 15, hb = lane >> 4;

  v16h afr = frag_ld(q16 + ((long)bp * NTOK + m0) * CDIM + h * HDIM, CDIM, lane);
  for (int ct = 0; ct < 16; ++ct) {
    v16h bfr =
        frag_ld(k16 + ((long)bp * NTOK + ct * 16) * CDIM + h * HDIM, CDIM, lane);
    v8f acc = {};
    acc = wmma16(afr, bfr, acc);
#pragma unroll
    for (int r = 0; r < 8; ++r) S[w][r + (hb << 3)][ct * 16 + n] = acc[r];
  }
  // softmax (scale 1/sqrt(32)) per row, whole wave cooperates per row
  const float scale = 0.17677669529663687f;
  for (int r = 0; r < 16; ++r) {
    float* row = &S[w][r][0];
    float vals[8];
    float mx = -3.4e38f;
#pragma unroll
    for (int j = 0; j < 8; ++j) {
      vals[j] = row[lane * 8 + j] * scale;
      mx = fmaxf(mx, vals[j]);
    }
    mx = wave_max(mx);
    float sum = 0.f;
#pragma unroll
    for (int j = 0; j < 8; ++j) {
      vals[j] = __expf(vals[j] - mx);
      sum += vals[j];
    }
    sum = wave_sum(sum);
    float inv = 1.f / sum;
#pragma unroll
    for (int j = 0; j < 8; ++j) row[lane * 8 + j] = vals[j] * inv;
  }
  // out = S @ vh  (vh via per-window transposed v: vt[bp][c][token])
  for (int ct = 0; ct < 2; ++ct) {
    v8f acc = {};
    for (int kk = 0; kk < 256; kk += 32) {
      const int m = lane & 15, c0 = (lane >> 4) << 3;
      const float* rp = &S[w][m][kk + c0];
      v4f f0 = *(const v4f*)(rp);
      v4f f1 = *(const v4f*)(rp + 4);
      v4f f2 = *(const v4f*)(rp + 16);
      v4f f3 = *(const v4f*)(rp + 20);
      v16h a;
#pragma unroll
      for (int j = 0; j < 4; ++j) {
        a[j] = (_Float16)f0[j];
        a[4 + j] = (_Float16)f1[j];
        a[8 + j] = (_Float16)f2[j];
        a[12 + j] = (_Float16)f3[j];
      }
      v16h b = frag_ld(
          vt16 + ((long)bp * CDIM + h * HDIM + ct * 16) * NTOK + kk, NTOK, lane);
      acc = wmma16(a, b, acc);
    }
#pragma unroll
    for (int r = 0; r < 8; ++r) {
      int m = m0 + r + (hb << 3);
      out16[((long)bp * NTOK + m) * CDIM + h * HDIM + ct * 16 + n] =
          (_Float16)acc[r];
    }
  }
}

// ---------------- data movement / pointwise kernels ----------------

// weight transpose + f16:  Wt[n][k] = (f16) W[k][n]
__global__ void k_wt(const float* __restrict__ W, _Float16* __restrict__ Wt,
                     int K, int Nc) {
  int tid = blockIdx.x * blockDim.x + threadIdx.x;
  if (tid >= K * Nc) return;
  int nn = tid / K, kk = tid - nn * K;
  Wt[tid] = (_Float16)W[(long)kk * Nc + nn];
}

// window partition + channels-last + f16
__global__ void k_pack(const float* __restrict__ x, const float* __restrict__ s,
                       _Float16* __restrict__ xt16, _Float16* __restrict__ st16) {
  int tid = blockIdx.x * blockDim.x + threadIdx.x;  // [Bp*N*C)
  int c = tid & 255, nn = (tid >> 8) & 255, bp = tid >> 16;
  int b = bp >> 6, wy = (bp >> 3) & 7, wx = bp & 7;
  int iy = nn >> 4, ix = nn & 15;
  long xp = (((long)b * CDIM + c) * 128 + wy * 16 + iy) * 128 + wx * 16 + ix;
  xt16[tid] = (_Float16)x[xp];
  st16[tid] = (_Float16)s[xp];
}

// row softmax in place, 256 cols, one wave per row
__global__ __launch_bounds__(256) void k_softmax_rows(float* __restrict__ A,
                                                      int rows) {
  int row = blockIdx.x * 8 + (threadIdx.x >> 5);
  if (row >= rows) return;
  int lane = threadIdx.x & 31;
  float* r = A + (long)row * 256 + lane * 8;
  v4f a = *(const v4f*)(r);
  v4f b = *(const v4f*)(r + 4);
  float v[8];
#pragma unroll
  for (int j = 0; j < 4; ++j) { v[j] = a[j]; v[4 + j] = b[j]; }
  float mx = -3.4e38f;
#pragma unroll
  for (int j = 0; j < 8; ++j) mx = fmaxf(mx, v[j]);
  mx = wave_max(mx);
  float sum = 0.f;
#pragma unroll
  for (int j = 0; j < 8; ++j) { v[j] = __expf(v[j] - mx); sum += v[j]; }
  sum = wave_sum(sum);
  float inv = 1.f / sum;
#pragma unroll
  for (int j = 0; j < 8; ++j) r[j] = v[j] * inv;
}

// row stats {max, sumexp} for second row-softmax
__global__ __launch_bounds__(256) void k_rowstats(const float* __restrict__ A,
                                                  float* __restrict__ st,
                                                  int rows) {
  int row = blockIdx.x * 8 + (threadIdx.x >> 5);
  if (row >= rows) return;
  int lane = threadIdx.x & 31;
  const float* r = A + (long)row * 256 + lane * 8;
  float v[8];
#pragma unroll
  for (int j = 0; j < 8; ++j) v[j] = r[j];
  float mx = -3.4e38f;
#pragma unroll
  for (int j = 0; j < 8; ++j) mx = fmaxf(mx, v[j]);
  mx = wave_max(mx);
  float sum = 0.f;
#pragma unroll
  for (int j = 0; j < 8; ++j) sum += __expf(v[j] - mx);
  sum = wave_sum(sum);
  if (lane == 0) { st[row * 2] = mx; st[row * 2 + 1] = sum; }
}

// column stats per (bp, m)
__global__ void k_colstats(const float* __restrict__ A, float* __restrict__ st) {
  int tid = blockIdx.x * blockDim.x + threadIdx.x;  // [Bp*256)
  int bp = tid >> 8, m = tid & 255;
  const float* base = A + ((long)bp * 256) * 256 + m;
  float mx = -3.4e38f;
  for (int nn = 0; nn < 256; ++nn) mx = fmaxf(mx, base[(long)nn * 256]);
  float sum = 0.f;
  for (int nn = 0; nn < 256; ++nn) sum += __expf(base[(long)nn * 256] - mx);
  st[tid * 2] = mx;
  st[tid * 2 + 1] = sum;
}

// top-8 over column m of s = softmax_row * softmax_col; write idx_t[bp][k][m]
__global__ void k_topk(const float* __restrict__ A, const float* __restrict__ rs,
                       const float* __restrict__ cs, int* __restrict__ idx_t) {
  int tid = blockIdx.x * blockDim.x + threadIdx.x;  // [Bp*256)
  int bp = tid >> 8, m = tid & 255;
  float cmax = cs[tid * 2], cinv = 1.f / cs[tid * 2 + 1];
  float bestv[KTOP];
  int besti[KTOP];
#pragma unroll
  for (int k = 0; k < KTOP; ++k) { bestv[k] = -3.4e38f; besti[k] = 0; }
  for (int nn = 0; nn < 256; ++nn) {
    float a = A[((long)bp * 256 + nn) * 256 + m];
    int ri = (bp * 256 + nn) * 2;
    float val = __expf(a - rs[ri]) / rs[ri + 1] * __expf(a - cmax) * cinv;
    if (val > bestv[KTOP - 1]) {
      int p = KTOP - 1;
      while (p > 0 && val > bestv[p - 1]) {
        bestv[p] = bestv[p - 1];
        besti[p] = besti[p - 1];
        --p;
      }
      bestv[p] = val;
      besti[p] = nn;
    }
  }
#pragma unroll
  for (int k = 0; k < KTOP; ++k)
    idx_t[((long)bp * KTOP + k) * 256 + m] = besti[k];
}

// LayerNorm -> f16 into right half of cat[row,512]; copy left half from src16
__global__ __launch_bounds__(256) void k_ln_cat(
    const float* __restrict__ in32, const float* __restrict__ g,
    const float* __restrict__ be, const _Float16* __restrict__ left16,
    _Float16* __restrict__ cat16, int rows) {
  int row = blockIdx.x * 8 + (threadIdx.x >> 5);
  if (row >= rows) return;
  int lane = threadIdx.x & 31;
  const float* r = in32 + (long)row * 256 + lane * 8;
  float v[8];
  float s = 0.f, s2 = 0.f;
#pragma unroll
  for (int j = 0; j < 8; ++j) { v[j] = r[j]; s += v[j]; s2 += v[j] * v[j]; }
  s = wave_sum(s);
  s2 = wave_sum(s2);
  float mu = s * (1.f / 256.f);
  float var = s2 * (1.f / 256.f) - mu * mu;
  float inv = rsqrtf(var + 1e-5f);
  v8h outp;
#pragma unroll
  for (int j = 0; j < 8; ++j) {
    int c = lane * 8 + j;
    outp[j] = (_Float16)((v[j] - mu) * inv * g[c] + be[c]);
  }
  *(v8h*)(cat16 + (long)row * 512 + 256 + lane * 8) = outp;
  v8h l = *(const v8h*)(left16 + (long)row * 256 + lane * 8);
  *(v8h*)(cat16 + (long)row * 512 + lane * 8) = l;
}

// out32 = res32 + LN(in32)
__global__ __launch_bounds__(256) void k_ln_res(
    const float* __restrict__ in32, const float* __restrict__ g,
    const float* __restrict__ be, const float* __restrict__ res32,
    float* __restrict__ out32, int rows) {
  int row = blockIdx.x * 8 + (threadIdx.x >> 5);
  if (row >= rows) return;
  int lane = threadIdx.x & 31;
  const float* r = in32 + (long)row * 256 + lane * 8;
  float v[8];
  float s = 0.f, s2 = 0.f;
#pragma unroll
  for (int j = 0; j < 8; ++j) { v[j] = r[j]; s += v[j]; s2 += v[j] * v[j]; }
  s = wave_sum(s);
  s2 = wave_sum(s2);
  float mu = s * (1.f / 256.f);
  float inv = rsqrtf(s2 * (1.f / 256.f) - mu * mu + 1e-5f);
#pragma unroll
  for (int j = 0; j < 8; ++j) {
    int c = lane * 8 + j;
    out32[(long)row * 256 + c] =
        res32[(long)row * 256 + c] + (v[j] - mu) * inv * g[c] + be[c];
  }
}

// winout = xt (gathered from x) + LN(in32)
__global__ __launch_bounds__(256) void k_ln_res_win(
    const float* __restrict__ in32, const float* __restrict__ g,
    const float* __restrict__ be, const float* __restrict__ x,
    float* __restrict__ out32, int rows) {
  int row = blockIdx.x * 8 + (threadIdx.x >> 5);
  if (row >= rows) return;
  int lane = threadIdx.x & 31;
  const float* r = in32 + (long)row * 256 + lane * 8;
  float v[8];
  float s = 0.f, s2 = 0.f;
#pragma unroll
  for (int j = 0; j < 8; ++j) { v[j] = r[j]; s += v[j]; s2 += v[j] * v[j]; }
  s = wave_sum(s);
  s2 = wave_sum(s2);
  float mu = s * (1.f / 256.f);
  float inv = rsqrtf(s2 * (1.f / 256.f) - mu * mu + 1e-5f);
  int bp = row >> 8, nn = row & 255;
  int b = bp >> 6, wy = (bp >> 3) & 7, wx = bp & 7;
  int iy = nn >> 4, ix = nn & 15;
#pragma unroll
  for (int j = 0; j < 8; ++j) {
    int c = lane * 8 + j;
    long xp = (((long)b * CDIM + c) * 128 + wy * 16 + iy) * 128 + wx * 16 + ix;
    out32[(long)row * 256 + c] = x[xp] + (v[j] - mu) * inv * g[c] + be[c];
  }
}

// topk gather: la_x[bp*8+kk][c] = winout[bp][kk][ idx_t[bp][kk][c] ]
__global__ void k_gather(const float* __restrict__ winout,
                         const int* __restrict__ idx_t,
                         float* __restrict__ lax32,
                         _Float16* __restrict__ lax16) {
  int tid = blockIdx.x * blockDim.x + threadIdx.x;  // [Bp*8*256)
  int c = tid & 255, kk = (tid >> 8) & 7, bp = tid >> 11;
  int nidx = idx_t[((long)bp * KTOP + kk) * 256 + c];
  float vv = winout[((long)bp * NTOK + kk) * 256 + nidx];
  lax32[tid] = vv;
  lax16[tid] = (_Float16)vv;
}

__global__ void k_elu1(float* __restrict__ a, int n) {
  int i = blockIdx.x * blockDim.x + threadIdx.x;
  if (i >= n) return;
  float x = a[i];
  a[i] = (x > 0.f) ? (x + 1.f) : __expf(x);
}
__global__ void k_scale(float* __restrict__ a, int n, float s) {
  int i = blockIdx.x * blockDim.x + threadIdx.x;
  if (i < n) a[i] *= s;
}

// KV[b][h][d][v] = sum_s K*V ; Ksum[b][h][d] = sum_s K
__global__ void k_kv(const float* __restrict__ Ke, const float* __restrict__ Ve,
                     float* __restrict__ KV, float* __restrict__ Ksum) {
  int tid = blockIdx.x * blockDim.x + threadIdx.x;  // [B*H*32*32)
  int vv = tid & 31, d = (tid >> 5) & 31, h = (tid >> 10) & 7, b = tid >> 13;
  float acc = 0.f, ks = 0.f;
  for (int s = 0; s < LSEQ; ++s) {
    float kd = Ke[((long)b * LSEQ + s) * 256 + h * HDIM + d];
    acc += kd * Ve[((long)b * LSEQ + s) * 256 + h * HDIM + vv];
    ks += kd;
  }
  KV[tid] = acc;
  if (vv == 0) Ksum[((long)b * NHEAD + h) * HDIM + d] = ks;
}

// lmsg[b][l][h*32+v] = (sum_d Q*KV) * z * L  (f16 for next GEMM)
__global__ void k_lmsg(const float* __restrict__ Qe, const float* __restrict__ KV,
                       const float* __restrict__ Ksum,
                       _Float16* __restrict__ lmsg16) {
  int tid = blockIdx.x * blockDim.x + threadIdx.x;  // [B*L*H)
  int h = tid & 7, l = (tid >> 3) & 511, b = tid >> 12;
  float q[HDIM];
  float z = 0.f;
#pragma unroll
  for (int d = 0; d < HDIM; ++d) {
    q[d] = Qe[((long)b * LSEQ + l) * 256 + h * HDIM + d];
    z += q[d] * Ksum[((long)b * NHEAD + h) * HDIM + d];
  }
  z = 1.f / (z + 1e-6f);
  const float* kvb = KV + ((long)b * NHEAD + h) * HDIM * HDIM;
  for (int vv = 0; vv < HDIM; ++vv) {
    float acc = 0.f;
#pragma unroll
    for (int d = 0; d < HDIM; ++d) acc += q[d] * kvb[d * HDIM + vv];
    lmsg16[((long)b * LSEQ + l) * 256 + h * HDIM + vv] =
        (_Float16)(acc * z * (float)LSEQ);
  }
}

__global__ void k_zero(float* __restrict__ p, long n) {
  long i = (long)blockIdx.x * blockDim.x + threadIdx.x;
  long stride = (long)gridDim.x * blockDim.x;
  for (; i < n; i += stride) p[i] = 0.f;
}

// mask[bp][idx][c] = laout[bp*8+kk][c]
__global__ void k_scatter(const float* __restrict__ laout,
                          const int* __restrict__ idx_t,
                          float* __restrict__ mask) {
  int tid = blockIdx.x * blockDim.x + threadIdx.x;  // [Bp*8*256)
  int c = tid & 255, kk = (tid >> 8) & 7, bp = tid >> 11;
  int nidx = idx_t[((long)bp * KTOP + kk) * 256 + c];
  mask[((long)bp * NTOK + nidx) * 256 + c] = laout[(long)tid];
}

// window reverse + add mask -> out[B,C,H,W]
__global__ void k_final(const float* __restrict__ winout,
                        const float* __restrict__ mask,
                        float* __restrict__ out) {
  int tid = blockIdx.x * blockDim.x + threadIdx.x;  // [B*C*H*W)
  int ww = tid & 127, hh = (tid >> 7) & 127, c = (tid >> 14) & 255, b = tid >> 22;
  int bp = b * 64 + (hh >> 4) * 8 + (ww >> 4);
  int nn = (hh & 15) * 16 + (ww & 15);
  long o = ((long)bp * NTOK + nn) * 256 + c;
  out[tid] = winout[o] + mask[o];
}

// ---------------- host orchestration ----------------

extern "C" void kernel_launch(void* const* d_in, const int* in_sizes, int n_in,
                              void* d_out, int out_size, void* d_ws,
                              size_t ws_size, hipStream_t stream) {
  const float* x      = (const float*)d_in[0];
  const float* source = (const float*)d_in[1];
  const float* m_wq = (const float*)d_in[2];
  const float* m_bq = (const float*)d_in[3];
  const float* m_wk = (const float*)d_in[4];
  const float* m_bk = (const float*)d_in[5];
  const float* m_wv = (const float*)d_in[6];
  const float* m_bv = (const float*)d_in[7];
  const float* m_wm = (const float*)d_in[8];
  const float* m_w1 = (const float*)d_in[9];
  const float* m_w2 = (const float*)d_in[10];
  const float* m_g1 = (const float*)d_in[11];
  const float* m_be1 = (const float*)d_in[12];
  const float* m_g2 = (const float*)d_in[13];
  const float* m_be2 = (const float*)d_in[14];
  const float* e_wq = (const float*)d_in[15];
  const float* e_wk = (const float*)d_in[16];
  const float* e_wv = (const float*)d_in[17];
  const float* e_wm = (const float*)d_in[18];
  const float* e_w1 = (const float*)d_in[19];
  const float* e_w2 = (const float*)d_in[20];
  const float* e_g1 = (const float*)d_in[21];
  const float* e_b1 = (const float*)d_in[22];
  const float* e_g2 = (const float*)d_in[23];
  const float* e_b2 = (const float*)d_in[24];
  float* out = (float*)d_out;

  // workspace bump allocator
  char* wsp = (char*)d_ws;
  size_t off = 0;
  auto alloc = [&](size_t bytes) -> void* {
    void* p = wsp + off;
    off = (off + bytes + 255) & ~(size_t)255;
    return p;
  };
  const long EWIN = (long)BPW * NTOK * CDIM;  // 16,777,216

  _Float16* xt16 = (_Float16*)alloc(EWIN * 2);
  _Float16* st16 = (_Float16*)alloc(EWIN * 2);
  _Float16* q16  = (_Float16*)alloc(EWIN * 2);
  _Float16* k16  = (_Float16*)alloc(EWIN * 2);
  _Float16* vt16 = (_Float16*)alloc(EWIN * 2);
  float* attn    = (float*)alloc(EWIN * 4);
  _Float16* ao16 = (_Float16*)alloc(EWIN * 2);
  float* tmp32   = (float*)alloc(EWIN * 4);
  _Float16* cat16 = (_Float16*)alloc((long)65536 * 512 * 2);
  _Float16* h116  = (_Float16*)alloc((long)65536 * 512 * 2);
  float* winout  = (float*)alloc(EWIN * 4);
  float* mask    = (float*)alloc(EWIN * 4);
  float* rstats  = (float*)alloc((long)65536 * 2 * 4);
  float* cstats  = (float*)alloc((long)65536 * 2 * 4);
  int* idx_t     = (int*)alloc((long)BPW * KTOP * 256 * 4);
  float* lax32   = (float*)alloc((long)2048 * 256 * 4);
  _Float16* lax16 = (_Float16*)alloc((long)2048 * 256 * 2);
  float* eQ = (float*)alloc((long)2048 * 256 * 4);
  float* eK = (float*)alloc((long)2048 * 256 * 4);
  float* eV = (float*)alloc((long)2048 * 256 * 4);
  float* KV = (float*)alloc((long)BB * NHEAD * HDIM * HDIM * 4);
  float* Ksum = (float*)alloc((long)BB * NHEAD * HDIM * 4);
  _Float16* lmsg16 = (_Float16*)alloc((long)2048 * 256 * 2);
  float* etmp32 = (float*)alloc((long)2048 * 256 * 4);
  _Float16* ecat16 = (_Float16*)alloc((long)2048 * 512 * 2);
  _Float16* eh16 = (_Float16*)alloc((long)2048 * 512 * 2);
  float* laout32 = (float*)alloc((long)2048 * 256 * 4);
  _Float16* wqT = (_Float16*)alloc(256 * 256 * 2);
  _Float16* wkT = (_Float16*)alloc(256 * 256 * 2);
  _Float16* wvT = (_Float16*)alloc(256 * 256 * 2);
  _Float16* wmT = (_Float16*)alloc(256 * 256 * 2);
  _Float16* w1T = (_Float16*)alloc(512 * 512 * 2);
  _Float16* w2T = (_Float16*)alloc(256 * 512 * 2);
  _Float16* ewqT = (_Float16*)alloc(256 * 256 * 2);
  _Float16* ewkT = (_Float16*)alloc(256 * 256 * 2);
  _Float16* ewvT = (_Float16*)alloc(256 * 256 * 2);
  _Float16* ewmT = (_Float16*)alloc(256 * 256 * 2);
  _Float16* ew1T = (_Float16*)alloc(512 * 512 * 2);
  _Float16* ew2T = (_Float16*)alloc(256 * 512 * 2);

  // 1) weight transposes to f16
  k_wt<<<256, 256, 0, stream>>>(m_wq, wqT, 256, 256);
  k_wt<<<256, 256, 0, stream>>>(m_wk, wkT, 256, 256);
  k_wt<<<256, 256, 0, stream>>>(m_wv, wvT, 256, 256);
  k_wt<<<256, 256, 0, stream>>>(m_wm, wmT, 256, 256);
  k_wt<<<1024, 256, 0, stream>>>(m_w1, w1T, 512, 512);
  k_wt<<<512, 256, 0, stream>>>(m_w2, w2T, 512, 256);
  k_wt<<<256, 256, 0, stream>>>(e_wq, ewqT, 256, 256);
  k_wt<<<256, 256, 0, stream>>>(e_wk, ewkT, 256, 256);
  k_wt<<<256, 256, 0, stream>>>(e_wv, ewvT, 256, 256);
  k_wt<<<256, 256, 0, stream>>>(e_wm, ewmT, 256, 256);
  k_wt<<<1024, 256, 0, stream>>>(e_w1, ew1T, 512, 512);
  k_wt<<<512, 256, 0, stream>>>(e_w2, ew2T, 512, 256);

  // 2) window partition
  k_pack<<<65536, 256, 0, stream>>>(x, source, xt16, st16);

  const int M = 65536;   // Bp*N rows
  const int ME = 2048;   // encoder rows

  // 3) q, k, v projections (WMMA, async-LDS staged B)
  k_gemm<<<dim3(4, 512), 256, 0, stream>>>(xt16, wqT, m_bq, nullptr, q16,
                                           nullptr, M, 256, 256, 0, 256, 0);
  k_gemm<<<dim3(4, 512), 256, 0, stream>>>(st16, wkT, m_bk, nullptr, k16,
                                           nullptr, M, 256, 256, 0, 256, 0);
  k_gemm<<<dim3(4, 512), 256, 0, stream>>>(st16, wvT, m_bv, nullptr, nullptr,
                                           vt16, M, 256, 256, 0, 256, 0);

  // 4) full-rank score q @ k^T per window (B = k16 rows, per-window stride)
  k_gemm<<<dim3(4, 512), 256, 0, stream>>>(q16, k16, nullptr, attn, nullptr,
                                           nullptr, M, 256, 256,
                                           (long)NTOK * CDIM, 256, 0);
  k_softmax_rows<<<8192, 256, 0, stream>>>(attn, 65536);

  // 5) fused per-head attention (WMMA + LDS softmax)
  k_headattn<<<8192, 128, 0, stream>>>(q16, k16, vt16, ao16);

  // 6) merge proj + LN -> cat, FFN, LN + residual -> window_output
  k_gemm<<<dim3(4, 512), 256, 0, stream>>>(ao16, wmT, nullptr, tmp32, nullptr,
                                           nullptr, M, 256, 256, 0, 256, 0);
  k_ln_cat<<<8192, 256, 0, stream>>>(tmp32, m_g1, m_be1, xt16, cat16, M);
  k_gemm<<<dim3(8, 512), 256, 0, stream>>>(cat16, w1T, nullptr, nullptr, h116,
                                           nullptr, M, 512, 512, 0, 256, 1);
  k_gemm<<<dim3(4, 512), 256, 0, stream>>>(h116, w2T, nullptr, tmp32, nullptr,
                                           nullptr, M, 256, 512, 0, 256, 0);
  k_ln_res_win<<<8192, 256, 0, stream>>>(tmp32, m_g2, m_be2, x, winout, M);

  // 7) double-softmax stats + top-k indices
  k_rowstats<<<8192, 256, 0, stream>>>(attn, rstats, 65536);
  k_colstats<<<256, 256, 0, stream>>>(attn, cstats);
  k_topk<<<256, 256, 0, stream>>>(attn, rstats, cstats, idx_t);

  // 8) gather -> encoder linear attention
  k_gather<<<2048, 256, 0, stream>>>(winout, idx_t, lax32, lax16);
  k_gemm<<<dim3(4, 16), 256, 0, stream>>>(lax16, ewqT, nullptr, eQ, nullptr,
                                          nullptr, ME, 256, 256, 0, 256, 0);
  k_gemm<<<dim3(4, 16), 256, 0, stream>>>(lax16, ewkT, nullptr, eK, nullptr,
                                          nullptr, ME, 256, 256, 0, 256, 0);
  k_gemm<<<dim3(4, 16), 256, 0, stream>>>(lax16, ewvT, nullptr, eV, nullptr,
                                          nullptr, ME, 256, 256, 0, 256, 0);
  k_elu1<<<2048, 256, 0, stream>>>(eQ, 2048 * 256);
  k_elu1<<<2048, 256, 0, stream>>>(eK, 2048 * 256);
  k_scale<<<2048, 256, 0, stream>>>(eV, 2048 * 256, 1.f / (float)LSEQ);
  k_kv<<<128, 256, 0, stream>>>(eK, eV, KV, Ksum);
  k_lmsg<<<64, 256, 0, stream>>>(eQ, KV, Ksum, lmsg16);
  k_gemm<<<dim3(4, 16), 256, 0, stream>>>(lmsg16, ewmT, nullptr, etmp32,
                                          nullptr, nullptr, ME, 256, 256, 0,
                                          256, 0);
  k_ln_cat<<<256, 256, 0, stream>>>(etmp32, e_g1, e_b1, lax16, ecat16, ME);
  k_gemm<<<dim3(8, 16), 256, 0, stream>>>(ecat16, ew1T, nullptr, nullptr, eh16,
                                          nullptr, ME, 512, 512, 0, 256, 1);
  k_gemm<<<dim3(4, 16), 256, 0, stream>>>(eh16, ew2T, nullptr, etmp32, nullptr,
                                          nullptr, ME, 256, 512, 0, 256, 0);
  k_ln_res<<<256, 256, 0, stream>>>(etmp32, e_g2, e_b2, lax32, laout32, ME);

  // 9) scatter refined tokens back, window reverse
  k_zero<<<4096, 256, 0, stream>>>(mask, EWIN);
  k_scatter<<<2048, 256, 0, stream>>>(laout32, idx_t, mask);
  k_final<<<65536, 256, 0, stream>>>(winout, mask, out);

  (void)in_sizes; (void)n_in; (void)out_size; (void)ws_size;
}
